// SelfAttention_69243462746222
// MI455X (gfx1250) — matrix operand order repaired and checked by reading, not compiled, not run on hardware
//
#include <hip/hip_runtime.h>
#include <hip/hip_bf16.h>
#include <math.h>

// ---------- types for WMMA ----------
typedef __attribute__((ext_vector_type(16))) __bf16 v16bf;
typedef __attribute__((ext_vector_type(8)))  float  f32x8;

union Frag {                 // 16 bf16 = 8 dwords, A or B fragment
    unsigned int u[8];
    v16bf v;
};

__device__ __forceinline__ f32x8 zero8() {
    f32x8 z;
    #pragma unroll
    for (int i = 0; i < 8; i++) z[i] = 0.0f;
    return z;
}

__device__ __forceinline__ unsigned int pack_bf16(float a, float b) {
    unsigned int ua = __float_as_uint(a);
    unsigned int ub = __float_as_uint(b);
    ua = (ua + 0x7FFFu + ((ua >> 16) & 1u)) >> 16;
    ub = (ub + 0x7FFFu + ((ub >> 16) & 1u)) >> 16;
    return ua | (ub << 16);
}
__device__ __forceinline__ unsigned short bf16_1(float a) {
    unsigned int ua = __float_as_uint(a);
    return (unsigned short)((ua + 0x7FFFu + ((ua >> 16) & 1u)) >> 16);
}

// ---------------------------------------------------------------------------
// Spectral-norm scalar chain: inv_sigma[w] = 1/sigma for W in {Wq,Wk,Wv,Wo}
//   t = W^T u ; v = t/(||t||+eps) ; w = W v ; sigma = ||w||^2/(||w||+eps)
// ---------------------------------------------------------------------------
#define SN_EPS 1e-12f

__global__ void sn_matvec_t(const float* __restrict__ Wq, const float* __restrict__ Wk,
                            const float* __restrict__ Wv, const float* __restrict__ Wo,
                            const float* __restrict__ uq, const float* __restrict__ uk,
                            const float* __restrict__ uv, const float* __restrict__ uo,
                            float* __restrict__ tvec) {
    const int wi = blockIdx.y;
    const float* W = (wi == 0) ? Wq : (wi == 1) ? Wk : (wi == 2) ? Wv : Wo;
    const float* u = (wi == 0) ? uq : (wi == 1) ? uk : (wi == 2) ? uv : uo;
    const int j = blockIdx.x * 256 + threadIdx.x;        // column of W
    float s = 0.0f;
    for (int i = 0; i < 1024; i++) s += W[i * 1024 + j] * u[i];
    tvec[wi * 1024 + j] = s;
}

__global__ void sn_norm(const float* __restrict__ tvec, float* __restrict__ nrm) {
    const int wi = blockIdx.x;
    __shared__ float red[8];
    float s = 0.0f;
    for (int i = threadIdx.x; i < 1024; i += 256) {
        float v = tvec[wi * 1024 + i];
        s += v * v;
    }
    #pragma unroll
    for (int off = 16; off >= 1; off >>= 1) s += __shfl_xor(s, off, 32);
    if ((threadIdx.x & 31) == 0) red[threadIdx.x >> 5] = s;
    __syncthreads();
    if (threadIdx.x == 0) {
        float t = 0.0f;
        #pragma unroll
        for (int i = 0; i < 8; i++) t += red[i];
        nrm[wi] = sqrtf(t);
    }
}

__global__ void sn_wv(const float* __restrict__ Wq, const float* __restrict__ Wk,
                      const float* __restrict__ Wv, const float* __restrict__ Wo,
                      const float* __restrict__ tvec, const float* __restrict__ nrm,
                      float* __restrict__ wvec) {
    const int wi = blockIdx.y;
    const float* W = (wi == 0) ? Wq : (wi == 1) ? Wk : (wi == 2) ? Wv : Wo;
    const int i = blockIdx.x * 256 + threadIdx.x;        // row of W
    const float inv = 1.0f / (nrm[wi] + SN_EPS);
    const float* t = tvec + wi * 1024;
    float s = 0.0f;
    for (int j = 0; j < 1024; j++) s += W[i * 1024 + j] * t[j];
    wvec[wi * 1024 + i] = s * inv;                       // = (W v)[i]
}

__global__ void sn_sigma(const float* __restrict__ wvec, float* __restrict__ inv_sigma) {
    const int wi = blockIdx.x;
    __shared__ float red[8];
    float s = 0.0f;
    for (int i = threadIdx.x; i < 1024; i += 256) {
        float v = wvec[wi * 1024 + i];
        s += v * v;
    }
    #pragma unroll
    for (int off = 16; off >= 1; off >>= 1) s += __shfl_xor(s, off, 32);
    if ((threadIdx.x & 31) == 0) red[threadIdx.x >> 5] = s;
    __syncthreads();
    if (threadIdx.x == 0) {
        float s2 = 0.0f;
        #pragma unroll
        for (int i = 0; i < 8; i++) s2 += red[i];
        float sigma = s2 / (sqrtf(s2) + SN_EPS);
        inv_sigma[wi] = 1.0f / sigma;
    }
}

// ---------------------------------------------------------------------------
// Projection GEMM: Y[m,n] = scale * sum_k A[m,k] * W[n,k]
//   M=4096 (b*L), N=K=1024. WG = 128 threads = 4 waves; tile 64(M) x 64(N).
//   Each wave: 16(M) x 64(N) = 4 accumulators of 16x16, K staged in LDS bf16.
//   ROPE: wave N-range is exactly one head (64) -> rotate_half pairs tile t
//   with t+2 in the same lane (register-only).
// ---------------------------------------------------------------------------
#define LD_STRIDE 40   // halves per LDS row (32 data + pad): conflict-free frag reads

template <bool A_BF16, bool ROPE, bool TRANS_OUT, bool OUT_F32>
__global__ __launch_bounds__(128) void proj_kernel(const void* __restrict__ Aptr,
                                                   const float* __restrict__ W,
                                                   const float* __restrict__ inv_sigma_p,
                                                   float extra_scale,
                                                   void* __restrict__ Out) {
    __shared__ unsigned short lA[64 * LD_STRIDE];
    __shared__ unsigned short lB[64 * LD_STRIDE];

    const int tid = threadIdx.x;
    const int w = tid >> 5, lane = tid & 31, lr = lane & 15, hi = lane >> 4;
    const int m0 = blockIdx.y * 64;
    const int n0 = blockIdx.x * 64;
    const float scale = inv_sigma_p[0] * extra_scale;

    f32x8 acc[4];
    #pragma unroll
    for (int t = 0; t < 4; t++) acc[t] = zero8();

    for (int k0 = 0; k0 < 1024; k0 += 32) {
        __syncthreads();
        // ---- stage A tile (64x32) as bf16 into LDS ----
        if constexpr (!A_BF16) {
            const float* A = (const float*)Aptr;
            #pragma unroll
            for (int p = 0; p < 4; p++) {
                int row = p * 16 + (tid >> 3);
                int c4 = (tid & 7) * 4;
                float4 va = *reinterpret_cast<const float4*>(A + (size_t)(m0 + row) * 1024 + k0 + c4);
                unsigned int* dst = (unsigned int*)(&lA[row * LD_STRIDE + c4]);
                dst[0] = pack_bf16(va.x, va.y);
                dst[1] = pack_bf16(va.z, va.w);
            }
        } else {
            const unsigned short* A = (const unsigned short*)Aptr;
            #pragma unroll
            for (int p = 0; p < 2; p++) {
                int row = p * 32 + (tid >> 2);
                int c8 = (tid & 3) * 8;
                uint4 va = *reinterpret_cast<const uint4*>(A + (size_t)(m0 + row) * 1024 + k0 + c8);
                unsigned int* dst = (unsigned int*)(&lA[row * LD_STRIDE + c8]);
                dst[0] = va.x; dst[1] = va.y; dst[2] = va.z; dst[3] = va.w;
            }
        }
        // ---- stage W tile (rows n0..n0+63, 32 k) as bf16 ----
        #pragma unroll
        for (int p = 0; p < 4; p++) {
            int row = p * 16 + (tid >> 3);
            int c4 = (tid & 7) * 4;
            float4 vb = *reinterpret_cast<const float4*>(W + (size_t)(n0 + row) * 1024 + k0 + c4);
            unsigned int* dst = (unsigned int*)(&lB[row * LD_STRIDE + c4]);
            dst[0] = pack_bf16(vb.x, vb.y);
            dst[1] = pack_bf16(vb.z, vb.w);
        }
        __syncthreads();

        // ---- A fragment: row = lane%16 (+wave offset), K pairs per §7.12.2 ----
        Frag af;
        const int arow = w * 16 + lr;
        #pragma unroll
        for (int v = 0; v < 8; v++) {
            int kp = 2 * (v & 3) + (hi ? 8 : 0) + 16 * (v >> 2);
            af.u[v] = *(const unsigned int*)(&lA[arow * LD_STRIDE + kp]);
        }
        #pragma unroll
        for (int t = 0; t < 4; t++) {
            Frag bf;
            const int brow = t * 16 + lr;     // output column n0 + brow
            #pragma unroll
            for (int v = 0; v < 8; v++) {
                int kp = 2 * (v & 3) + (hi ? 8 : 0) + 16 * (v >> 2);
                bf.u[v] = *(const unsigned int*)(&lB[brow * LD_STRIDE + kp]);
            }
            acc[t] = __builtin_amdgcn_wmma_f32_16x16x32_bf16(false, af.v, false, bf.v,
                                                             (short)0, acc[t], false, false);
        }
    }

    // ---- epilogue: scale, optional RoPE, store ----
    #pragma unroll
    for (int t = 0; t < 4; t++)
        #pragma unroll
        for (int r = 0; r < 8; r++) acc[t][r] *= scale;

    if constexpr (ROPE) {
        const float lg = -logf(10000.0f) / 32.0f;
        const float e0 = __expf(lg * (float)lr);          // inv_freq[lr]
        const float e1 = __expf(lg * (float)(16 + lr));   // inv_freq[16+lr]
        #pragma unroll
        for (int r = 0; r < 8; r++) {
            int srow = (m0 + w * 16 + r + hi * 8) & 2047;  // sequence position
            float c0, s0, c1, s1;
            __sincosf((float)srow * e0, &s0, &c0);
            __sincosf((float)srow * e1, &s1, &c1);
            float a0 = acc[0][r], a1 = acc[1][r], a2 = acc[2][r], a3 = acc[3][r];
            acc[0][r] = a0 * c0 - a2 * s0;   // j < 32:   x*cos - x[j+32]*sin
            acc[2][r] = a2 * c0 + a0 * s0;   // j >= 32:  x*cos + x[j-32]*sin
            acc[1][r] = a1 * c1 - a3 * s1;
            acc[3][r] = a3 * c1 + a1 * s1;
        }
    }

    #pragma unroll
    for (int t = 0; t < 4; t++) {
        #pragma unroll
        for (int r = 0; r < 8; r++) {
            int m = m0 + w * 16 + r + hi * 8;
            int n = n0 + t * 16 + lr;
            float val = acc[t][r];
            if constexpr (OUT_F32) {
                ((float*)Out)[(size_t)m * 1024 + n] = val;
            } else {
                int b = m >> 11, s = m & 2047;
                int h = n >> 6, dd = n & 63;
                size_t idx = TRANS_OUT
                                 ? ((size_t)((b * 16 + h) * 64 + dd)) * 2048 + s   // V^T [b,h,d,L]
                                 : ((size_t)((b * 16 + h) * 2048 + s)) * 64 + dd;  // Q/K [b,h,L,d]
                ((unsigned short*)Out)[idx] = bf16_1(val);
            }
        }
    }
}

// ---------------------------------------------------------------------------
// Flash attention: grid (L/64, heads, batch), 128 threads = 4 waves.
// Wave owns 16 query rows; loops keys in chunks of 64 (4 score tiles):
//   scores (16x64) = Q(16x64) @ K^T : 8 WMMAs
//   one online-softmax pass per 64 keys (width-16 shfl reductions)
//   P restaged via wave-private LDS (C-layout -> A-layout), out += P @ V : 8 WMMAs
// Next K chunk is prefetched (global_prefetch_b8) while softmax runs.
// ---------------------------------------------------------------------------
#define PSTRIDE 72   // halves per P row (64 data + pad)

__global__ __launch_bounds__(128) void attn_kernel(const unsigned short* __restrict__ q,
                                                   const unsigned short* __restrict__ k,
                                                   const unsigned short* __restrict__ vT,
                                                   const unsigned char* __restrict__ mask,
                                                   unsigned short* __restrict__ aout) {
    __shared__ unsigned short lP[4 * 16 * PSTRIDE];

    const int tid = threadIdx.x;
    const int w = tid >> 5, lane = tid & 31, lr = lane & 15, hi = lane >> 4;
    const int h = blockIdx.y, b = blockIdx.z;
    const int s0 = blockIdx.x * 64 + w * 16;
    const size_t bh = (size_t)(b * 16 + h);
    unsigned short* myP = &lP[w * 16 * PSTRIDE];
    const unsigned char* bmask = mask + b * 2048;

    // Q fragments for the two 32-wide d chunks (loop invariant)
    Frag qf[2];
    {
        const unsigned short* qbase = q + (bh * 2048 + (s0 + lr)) * 64;
        #pragma unroll
        for (int c = 0; c < 2; c++)
            #pragma unroll
            for (int v = 0; v < 8; v++) {
                int kp = c * 32 + 2 * (v & 3) + (hi ? 8 : 0) + 16 * (v >> 2);
                qf[c].u[v] = *(const unsigned int*)(qbase + kp);
            }
    }

    f32x8 acc[4];
    #pragma unroll
    for (int t = 0; t < 4; t++) acc[t] = zero8();
    float mrow[8], lrow[8];
    #pragma unroll
    for (int r = 0; r < 8; r++) { mrow[r] = -3.0e38f; lrow[r] = 0.0f; }

    for (int t0 = 0; t0 < 2048; t0 += 64) {
        // prefetch next K chunk rows into near cache while we work on this one
        {
            int tn = (t0 + 64) & 2047;
            __builtin_prefetch(k + (bh * 2048 + (tn + lr)) * 64, 0, 3);
            __builtin_prefetch(k + (bh * 2048 + (tn + 32 + lr)) * 64, 0, 3);
        }

        // ---- scores: four 16x16 tiles, K-dim=64 split into two WMMAs each ----
        f32x8 sc[4];
        bool mk[4];
        #pragma unroll
        for (int c = 0; c < 4; c++) {
            mk[c] = bmask[t0 + c * 16 + lr] != 0;
            const unsigned short* kbase = k + (bh * 2048 + (t0 + c * 16 + lr)) * 64;
            Frag kf0, kf1;
            #pragma unroll
            for (int v = 0; v < 8; v++) {
                int kp = 2 * (v & 3) + (hi ? 8 : 0) + 16 * (v >> 2);
                kf0.u[v] = *(const unsigned int*)(kbase + kp);
                kf1.u[v] = *(const unsigned int*)(kbase + 32 + kp);
            }
            f32x8 z = zero8();
            z = __builtin_amdgcn_wmma_f32_16x16x32_bf16(false, qf[0].v, false, kf0.v,
                                                        (short)0, z, false, false);
            sc[c] = __builtin_amdgcn_wmma_f32_16x16x32_bf16(false, qf[1].v, false, kf1.v,
                                                            (short)0, z, false, false);
        }
        // mask (key-position dependent only -> per-lane uniform across rows)
        #pragma unroll
        for (int c = 0; c < 4; c++)
            if (!mk[c]) {
                #pragma unroll
                for (int r = 0; r < 8; r++) sc[c][r] = -3.0e38f;
            }

        // ---- one online-softmax pass over 64 keys; write P (bf16) to LDS ----
        #pragma unroll
        for (int r = 0; r < 8; r++) {
            float rm = fmaxf(fmaxf(sc[0][r], sc[1][r]), fmaxf(sc[2][r], sc[3][r]));
            #pragma unroll
            for (int off = 8; off >= 1; off >>= 1) rm = fmaxf(rm, __shfl_xor(rm, off, 32));
            float nm = fmaxf(mrow[r], rm);
            float al = __expf(mrow[r] - nm);
            float p0 = __expf(sc[0][r] - nm);
            float p1 = __expf(sc[1][r] - nm);
            float p2 = __expf(sc[2][r] - nm);
            float p3 = __expf(sc[3][r] - nm);
            float rs = (p0 + p1) + (p2 + p3);
            #pragma unroll
            for (int off = 8; off >= 1; off >>= 1) rs += __shfl_xor(rs, off, 32);
            lrow[r] = lrow[r] * al + rs;
            mrow[r] = nm;
            acc[0][r] *= al; acc[1][r] *= al; acc[2][r] *= al; acc[3][r] *= al;
            int prow = r + hi * 8;
            myP[prow * PSTRIDE + lr]      = bf16_1(p0);
            myP[prow * PSTRIDE + 16 + lr] = bf16_1(p1);
            myP[prow * PSTRIDE + 32 + lr] = bf16_1(p2);
            myP[prow * PSTRIDE + 48 + lr] = bf16_1(p3);
        }

        // ---- P as A-fragments (DS ops are in-order per wave; no barrier) ----
        Frag pf[2];
        #pragma unroll
        for (int c = 0; c < 2; c++)
            #pragma unroll
            for (int v = 0; v < 8; v++) {
                int kp = c * 32 + 2 * (v & 3) + (hi ? 8 : 0) + 16 * (v >> 2);
                pf[c].u[v] = *(const unsigned int*)(&myP[lr * PSTRIDE + kp]);
            }
        // ---- out += P @ V  (V^T layout makes B pairs contiguous dwords) ----
        #pragma unroll
        for (int t = 0; t < 4; t++) {
            const unsigned short* vbase = vT + (bh * 64 + t * 16 + lr) * 2048 + t0;
            Frag vf0, vf1;
            #pragma unroll
            for (int v = 0; v < 8; v++) {
                int kp = 2 * (v & 3) + (hi ? 8 : 0) + 16 * (v >> 2);
                vf0.u[v] = *(const unsigned int*)(vbase + kp);
                vf1.u[v] = *(const unsigned int*)(vbase + 32 + kp);
            }
            acc[t] = __builtin_amdgcn_wmma_f32_16x16x32_bf16(false, pf[0].v, false, vf0.v,
                                                             (short)0, acc[t], false, false);
            acc[t] = __builtin_amdgcn_wmma_f32_16x16x32_bf16(false, pf[1].v, false, vf1.v,
                                                             (short)0, acc[t], false, false);
        }
    }

    // ---- normalize and store bf16 into [b*L, H] for the output projection ----
    #pragma unroll
    for (int r = 0; r < 8; r++) {
        float inv = 1.0f / lrow[r];
        int row = b * 2048 + s0 + r + hi * 8;
        #pragma unroll
        for (int t = 0; t < 4; t++) {
            aout[(size_t)row * 1024 + h * 64 + t * 16 + lr] = bf16_1(acc[t][r] * inv);
        }
    }
}

// ---------------------------------------------------------------------------
// launch
// ---------------------------------------------------------------------------
extern "C" void kernel_launch(void* const* d_in, const int* in_sizes, int n_in,
                              void* d_out, int out_size, void* d_ws, size_t ws_size,
                              hipStream_t stream) {
    (void)in_sizes; (void)n_in; (void)out_size; (void)ws_size;

    const float* x  = (const float*)d_in[0];
    const unsigned char* am = (const unsigned char*)d_in[1];
    const float* Wq = (const float*)d_in[2];
    const float* Wk = (const float*)d_in[3];
    const float* Wv = (const float*)d_in[4];
    const float* Wo = (const float*)d_in[5];
    const float* uq = (const float*)d_in[6];
    const float* uk = (const float*)d_in[7];
    const float* uv = (const float*)d_in[8];
    const float* uo = (const float*)d_in[9];

    char* ws = (char*)d_ws;
    float* inv_sigma = (float*)ws;                         // 4 floats
    float* nrm  = (float*)(ws + 256);                      // 4 floats
    float* tvec = (float*)(ws + 1024);                     // 4*1024 f32
    float* wvec = (float*)(ws + 20480);                    // 4*1024 f32
    const size_t ELEMS = (size_t)2 * 16 * 2048 * 64;       // 4 Mi elements
    unsigned short* qb   = (unsigned short*)(ws + 65536);
    unsigned short* kb   = qb + ELEMS;
    unsigned short* vT   = kb + ELEMS;
    unsigned short* aout = vT + ELEMS;

    // spectral-norm scalars
    sn_matvec_t<<<dim3(4, 4), 256, 0, stream>>>(Wq, Wk, Wv, Wo, uq, uk, uv, uo, tvec);
    sn_norm<<<4, 256, 0, stream>>>(tvec, nrm);
    sn_wv<<<dim3(4, 4), 256, 0, stream>>>(Wq, Wk, Wv, Wo, tvec, nrm, wvec);
    sn_sigma<<<4, 256, 0, stream>>>(wvec, inv_sigma);

    // projections (Q folds the 1/sqrt(d)=0.125 score scale)
    proj_kernel<false, true,  false, false><<<dim3(16, 64), 128, 0, stream>>>(x, Wq, inv_sigma + 0, 0.125f, qb);
    proj_kernel<false, true,  false, false><<<dim3(16, 64), 128, 0, stream>>>(x, Wk, inv_sigma + 1, 1.0f,   kb);
    proj_kernel<false, false, true,  false><<<dim3(16, 64), 128, 0, stream>>>(x, Wv, inv_sigma + 2, 1.0f,   vT);

    // attention
    attn_kernel<<<dim3(32, 16, 2), 128, 0, stream>>>(qb, kb, vT, am, aout);

    // output projection -> f32 d_out
    proj_kernel<true, false, false, true><<<dim3(16, 64), 128, 0, stream>>>(aout, Wo, inv_sigma + 3, 1.0f, (float*)d_out);
}
